// GatedLinearRecurrence_31937376813254
// MI455X (gfx1250) — matrix operand order repaired and checked by reading, not compiled, not run on hardware
//
#include <hip/hip_runtime.h>

// ---------------------------------------------------------------------------
// Problem constants (fixed by the reference): B=4, S=4096, D=H=1024
// ---------------------------------------------------------------------------
static constexpr int kB  = 4;
static constexpr int kS  = 4096;
static constexpr int kD  = 1024;
static constexpr int kH  = 1024;
static constexpr int kBS = kB * kS;          // 16384 tokens
static constexpr int kNC = 32;               // scan chunks per sequence
static constexpr int kCH = kS / kNC;         // 128 steps per chunk

typedef __attribute__((ext_vector_type(16))) __bf16 v16bf;
typedef __attribute__((ext_vector_type(8)))  __bf16 v8bf;
typedef __attribute__((ext_vector_type(4)))  __bf16 v4bf;
typedef __attribute__((ext_vector_type(8)))  float  v8f;

union AV { v16bf v; v8bf h[2]; };

// LDS byte offset of a __shared__ pointer: generic LDS addresses are
// {SHARED_BASE[63:32], offset[31:0]} on CDNA5, so truncation yields the offset.
__device__ __forceinline__ unsigned lds_offset(const void* p) {
    return (unsigned)(unsigned long long)(uintptr_t)p;
}

// ---------------------------------------------------------------------------
// fp32 -> bf16 conversion (vectorized 4/thread)
// ---------------------------------------------------------------------------
__global__ __launch_bounds__(256)
void f2bf_kernel(const float* __restrict__ in, __bf16* __restrict__ out, int n4) {
    int i = blockIdx.x * 256 + threadIdx.x;
    if (i < n4) {
        float4 v = ((const float4*)in)[i];
        v4bf o;
        o[0] = (__bf16)v.x; o[1] = (__bf16)v.y;
        o[2] = (__bf16)v.z; o[3] = (__bf16)v.w;
        ((v4bf*)out)[i] = o;
    }
}

// log_a[h] = log(sigmoid(lam[h]) + 1e-8)
__global__ __launch_bounds__(256)
void loga_kernel(const float* __restrict__ lam, float* __restrict__ loga) {
    int i = blockIdx.x * 256 + threadIdx.x;
    if (i < kH) {
        float a = 1.0f / (1.0f + __expf(-lam[i]));
        loga[i] = __logf(a + 1e-8f);
    }
}

// ---------------------------------------------------------------------------
// Fused 3-projection GEMM (x @ {W_in,W_ig,W_rg}^T) + gating epilogue.
// Block tile: 128 rows x 16 h-cols; 8 waves each own a 16-row subtile and
// SHARE the 3 B fragments, async-staged into LDS (double-buffered) with
// global_load_async_to_lds_b128 / s_wait_asynccnt.  Emits
// v_wmma_f32_16x16x32_bf16 + ds_load_b128 + async-load path.
// ---------------------------------------------------------------------------
__global__ __launch_bounds__(256)
void proj_gemm_kernel(const __bf16* __restrict__ xbf,   // [BS, D] row-major
                      const __bf16* __restrict__ wbf,   // [3, H, D] row-major
                      const float*  __restrict__ b_ig,
                      const float*  __restrict__ b_rg,
                      const float*  __restrict__ loga,
                      float* __restrict__ a_out,        // [BS, H]
                      float* __restrict__ g_out)        // [BS, H]
{
    // B stage: 3 matrices x 16 rows x 32 k of bf16 = 3 KB per buffer
    __shared__ __align__(16) __bf16 smemB[2][3 * 16 * 32];

    const int tidb  = threadIdx.x;
    const int wave  = tidb >> 5;
    const int lane  = tidb & 31;
    const int rb    = blockIdx.x >> 6;   // 128-row block  (BS/128 = 128)
    const int ct    = blockIdx.x & 63;   // 16-col tile    (H/16  = 64)
    const int m0    = rb * 128 + wave * 16;
    const int h0    = ct << 4;
    const int rl    = lane & 15;
    const int hi    = lane >> 4;
    const int ab    = hi << 4;           // A K-base offset (0 or 16)
    const int bb    = hi << 4;           // B K-base offset (0 or 16)

    const __bf16* arow = xbf + (size_t)(m0 + rl) * kD;

    // Async stage of one 32-wide K slice of all three B tiles into smemB[buf].
    // Threads 0..191 each move one 16-byte chunk (6 waves x 1 instruction).
    auto stageB = [&](int buf, int k) {
        if (tidb < 192) {
            const int mat = tidb >> 6;            // 0..2
            const int r   = (tidb >> 2) & 15;     // B row (h within tile)
            const int kp  = (tidb & 3) << 3;      // 0,8,16,24 (elements)
            const __bf16* gp = wbf + ((size_t)mat * kH + h0 + r) * kD + k + kp;
            unsigned lds = lds_offset(&smemB[buf][(mat * 16 + r) * 32 + kp]);
            asm volatile("global_load_async_to_lds_b128 %0, %1, off"
                         :: "v"(lds), "v"((unsigned long long)(uintptr_t)gp)
                         : "memory");
        }
    };

    v8f c0 = {}, c1 = {}, c2 = {};

    stageB(0, 0);
    asm volatile("s_wait_asynccnt 0x0" ::: "memory");
    __syncthreads();

    for (int kk = 0; kk < kD / 32; ++kk) {
        const int cur = kk & 1;
        if (kk + 1 < kD / 32)
            stageB(cur ^ 1, (kk + 1) * 32);

        const int k = kk * 32;
        AV a, bA, bBv, bC;
        a.h[0] = *(const v8bf*)(arow + k + ab);
        a.h[1] = *(const v8bf*)(arow + k + ab + 8);
        const __bf16* s0 = &smemB[cur][(0 * 16 + rl) * 32 + bb];
        const __bf16* s1 = &smemB[cur][(1 * 16 + rl) * 32 + bb];
        const __bf16* s2 = &smemB[cur][(2 * 16 + rl) * 32 + bb];
        bA.h[0]  = *(const v8bf*)(s0);
        bA.h[1]  = *(const v8bf*)(s0 + 8);
        bBv.h[0] = *(const v8bf*)(s1);
        bBv.h[1] = *(const v8bf*)(s1 + 8);
        bC.h[0]  = *(const v8bf*)(s2);
        bC.h[1]  = *(const v8bf*)(s2 + 8);

        c0 = __builtin_amdgcn_wmma_f32_16x16x32_bf16(false, a.v, false, bA.v,  (short)0, c0, false, false);
        c1 = __builtin_amdgcn_wmma_f32_16x16x32_bf16(false, a.v, false, bBv.v, (short)0, c1, false, false);
        c2 = __builtin_amdgcn_wmma_f32_16x16x32_bf16(false, a.v, false, bC.v,  (short)0, c2, false, false);

        asm volatile("s_wait_asynccnt 0x0" ::: "memory");
        __syncthreads();
    }

    // Gating epilogue: i=sig(ig), r=sig(rg), a=exp(8*r*log_a), g=sqrt(1-a^2+eps)*i*x_proj
    const int   hcol = h0 + rl;
    const float bi = b_ig[hcol], br = b_rg[hcol], la = loga[hcol];
    const int   r0 = m0 + (hi << 3);
#pragma unroll
    for (int j = 0; j < 8; ++j) {
        float pin = c0[j];
        float iv  = 1.0f / (1.0f + __expf(-(c1[j] + bi)));
        float rv  = 1.0f / (1.0f + __expf(-(c2[j] + br)));
        float av  = __expf(8.0f * rv * la);
        float gv  = sqrtf(1.0f - av * av + 1e-6f) * iv * pin;
        size_t idx = (size_t)(r0 + j) * kH + hcol;
        a_out[idx] = av;
        g_out[idx] = gv;
    }
}

// ---------------------------------------------------------------------------
// Chunked linear scan:  h_t = a_t * h_{t-1} + g_t
// pass1: per (b,h,chunk) -> (prod a, local end h);  pass2: combine carries;
// pass3: re-scan with carry-in, writing h in place of a.
// tid layout: (b*NC + c)*H + h  (consecutive h -> coalesced)
// ---------------------------------------------------------------------------
__global__ __launch_bounds__(256)
void scan_pass1(const float* __restrict__ a, const float* __restrict__ g,
                float* __restrict__ cA, float* __restrict__ cG)
{
    const int tid = blockIdx.x * 256 + threadIdx.x;
    const int h = tid & (kH - 1);
    const int c = (tid >> 10) & (kNC - 1);
    const int b = tid >> 15;
    size_t p = ((size_t)(b * kS + c * kCH)) * kH + h;
    float A = 1.0f, G = 0.0f;
    for (int t = 0; t < kCH; ++t) {
        __builtin_prefetch(a + p + (size_t)8 * kH, 0, 0);
        __builtin_prefetch(g + p + (size_t)8 * kH, 0, 0);
        float at = a[p], gt = g[p];
        A *= at;
        G = fmaf(at, G, gt);
        p += kH;
    }
    cA[tid] = A;
    cG[tid] = G;
}

__global__ __launch_bounds__(256)
void scan_pass2(const float* __restrict__ cA, const float* __restrict__ cG,
                float* __restrict__ cin)
{
    const int tid = blockIdx.x * 256 + threadIdx.x;   // b*H + h
    const int h = tid & (kH - 1);
    const int b = tid >> 10;
    float carry = 0.0f;                               // h0 == 0
    for (int c = 0; c < kNC; ++c) {
        size_t idx = ((size_t)(b * kNC + c) << 10) + h;
        cin[idx] = carry;
        carry = fmaf(cA[idx], carry, cG[idx]);
    }
}

__global__ __launch_bounds__(256)
void scan_pass3(float* __restrict__ ah,               // a in, h out (in place)
                const float* __restrict__ g,
                const float* __restrict__ cin)
{
    const int tid = blockIdx.x * 256 + threadIdx.x;
    const int h = tid & (kH - 1);
    const int c = (tid >> 10) & (kNC - 1);
    const int b = tid >> 15;
    size_t p = ((size_t)(b * kS + c * kCH)) * kH + h;
    float hv = cin[tid];
    for (int t = 0; t < kCH; ++t) {
        __builtin_prefetch(ah + p + (size_t)8 * kH, 0, 0);
        __builtin_prefetch(g  + p + (size_t)8 * kH, 0, 0);
        float at = ah[p], gt = g[p];
        hv = fmaf(at, hv, gt);
        ah[p] = hv;
        p += kH;
    }
}

// ---------------------------------------------------------------------------
// RMSNorm over H per token -> bf16 (fused scale by norm_w)
// ---------------------------------------------------------------------------
__global__ __launch_bounds__(256)
void rmsnorm_kernel(const float* __restrict__ hbuf, const float* __restrict__ nw,
                    __bf16* __restrict__ out)
{
    __shared__ float red[256];
    const int tok = blockIdx.x;
    const int t = threadIdx.x;
    const float4 v = *(const float4*)(hbuf + (size_t)tok * kH + t * 4);
    red[t] = v.x * v.x + v.y * v.y + v.z * v.z + v.w * v.w;
    __syncthreads();
    for (int o = 128; o > 0; o >>= 1) {
        if (t < o) red[t] += red[t + o];
        __syncthreads();
    }
    const float scale = rsqrtf(red[0] * (1.0f / kH) + 1e-6f);
    const float4 w = *(const float4*)(nw + t * 4);
    v4bf o;
    o[0] = (__bf16)(v.x * scale * w.x);
    o[1] = (__bf16)(v.y * scale * w.y);
    o[2] = (__bf16)(v.z * scale * w.z);
    o[3] = (__bf16)(v.w * scale * w.w);
    *(v4bf*)(out + (size_t)tok * kH + t * 4) = o;
}

// ---------------------------------------------------------------------------
// Output GEMM: out[bs,d] = sum_h normed[bs,h] * W_out[d,h]  (both K-major)
// ---------------------------------------------------------------------------
__global__ __launch_bounds__(256)
void out_gemm_kernel(const __bf16* __restrict__ nbf,  // [BS, H]
                     const __bf16* __restrict__ wo,   // [D, H]
                     float* __restrict__ out)         // [BS, D]
{
    const int wid     = (blockIdx.x * 256 + threadIdx.x) >> 5;
    const int lane    = threadIdx.x & 31;
    const int rowTile = wid >> 6;        // BS/16
    const int colTile = wid & 63;        // D/16
    const int m0 = rowTile << 4;
    const int d0 = colTile << 4;
    const int rl = lane & 15;
    const int hi = lane >> 4;
    const int ab = hi << 4;
    const int bb = hi << 4;

    const __bf16* arow = nbf + (size_t)(m0 + rl) * kH;
    const __bf16* brow = wo  + (size_t)(d0 + rl) * kH;

    v8f c = {};
#pragma unroll 2
    for (int k = 0; k < kH; k += 32) {
        AV a, b;
        a.h[0] = *(const v8bf*)(arow + k + ab);
        a.h[1] = *(const v8bf*)(arow + k + ab + 8);
        b.h[0] = *(const v8bf*)(brow + k + bb);
        b.h[1] = *(const v8bf*)(brow + k + bb + 8);
        c = __builtin_amdgcn_wmma_f32_16x16x32_bf16(false, a.v, false, b.v, (short)0, c, false, false);
    }

    const int dcol = d0 + rl;
    const int r0 = m0 + (hi << 3);
#pragma unroll
    for (int j = 0; j < 8; ++j)
        out[(size_t)(r0 + j) * kD + dcol] = c[j];
}

// ---------------------------------------------------------------------------
// Host launcher
// ---------------------------------------------------------------------------
extern "C" void kernel_launch(void* const* d_in, const int* in_sizes, int n_in,
                              void* d_out, int out_size, void* d_ws, size_t ws_size,
                              hipStream_t stream) {
    const float* x      = (const float*)d_in[0];
    const float* W_in   = (const float*)d_in[1];
    const float* W_ig   = (const float*)d_in[2];
    const float* b_ig   = (const float*)d_in[3];
    const float* W_rg   = (const float*)d_in[4];
    const float* b_rg   = (const float*)d_in[5];
    const float* lam    = (const float*)d_in[6];
    const float* W_out  = (const float*)d_in[7];
    // d_in[8] = h0 (zeros, folded into scan init); d_in[9] = norm_w
    const float* norm_w = (const float*)d_in[9];

    char* ws = (char*)d_ws;
    size_t off = 0;
    auto alloc = [&](size_t bytes) -> void* {
        void* p = ws + off;
        off = (off + bytes + 255) & ~(size_t)255;
        return p;
    };

    __bf16* xbf  = (__bf16*)alloc((size_t)kBS * kD * 2);   // 32 MB (reused for normed)
    __bf16* wbf  = (__bf16*)alloc((size_t)3 * kH * kD * 2);
    __bf16* wobf = (__bf16*)alloc((size_t)kD * kH * 2);
    float*  loga = (float*)alloc((size_t)kH * 4);
    float*  abuf = (float*)alloc((size_t)kBS * kH * 4);    // a_t, then h in place
    float*  gbuf = (float*)alloc((size_t)kBS * kH * 4);    // gated
    float*  cA   = (float*)alloc((size_t)kB * kNC * kH * 4);
    float*  cG   = (float*)alloc((size_t)kB * kNC * kH * 4);
    float*  cin  = (float*)alloc((size_t)kB * kNC * kH * 4);
    __bf16* nbf  = xbf;  // reuse: x-bf16 dead after projection GEMM

    // fp32 -> bf16
    {
        int n4 = (kBS * kD) / 4;
        f2bf_kernel<<<(n4 + 255) / 256, 256, 0, stream>>>(x, xbf, n4);
        int w4 = (kH * kD) / 4;
        f2bf_kernel<<<(w4 + 255) / 256, 256, 0, stream>>>(W_in, wbf, w4);
        f2bf_kernel<<<(w4 + 255) / 256, 256, 0, stream>>>(W_ig, wbf + (size_t)kH * kD, w4);
        f2bf_kernel<<<(w4 + 255) / 256, 256, 0, stream>>>(W_rg, wbf + (size_t)2 * kH * kD, w4);
        f2bf_kernel<<<(w4 + 255) / 256, 256, 0, stream>>>(W_out, wobf, w4);
    }
    loga_kernel<<<(kH + 255) / 256, 256, 0, stream>>>(lam, loga);

    // Fused projections + gating (WMMA + async-LDS staging)
    {
        int blocks = (kBS / 128) * (kH / 16);       // 128 * 64 = 8192
        proj_gemm_kernel<<<blocks, 256, 0, stream>>>(xbf, wbf, b_ig, b_rg, loga, abuf, gbuf);
    }

    // Chunked scan
    scan_pass1<<<(kB * kNC * kH) / 256, 256, 0, stream>>>(abuf, gbuf, cA, cG);
    scan_pass2<<<(kB * kH) / 256, 256, 0, stream>>>(cA, cG, cin);
    scan_pass3<<<(kB * kNC * kH) / 256, 256, 0, stream>>>(abuf, gbuf, cin);

    // RMSNorm -> bf16
    rmsnorm_kernel<<<kBS, 256, 0, stream>>>(abuf, norm_w, nbf);

    // Output projection (WMMA)
    {
        int waves = (kBS / 16) * (kD / 16);
        out_gemm_kernel<<<waves / 8, 256, 0, stream>>>(nbf, wobf, (float*)d_out);
    }
}
